// GAT_47210280518299
// MI455X (gfx1250) — matrix operand order repaired and checked by reading, not compiled, not run on hardware
//
#include <hip/hip_runtime.h>
#include <math.h>

typedef __attribute__((ext_vector_type(16))) _Float16 v16h;
typedef __attribute__((ext_vector_type(8)))  float    v8f;
typedef __attribute__((ext_vector_type(4)))  unsigned u32x4;
typedef __attribute__((ext_vector_type(8)))  int      i32x8;
typedef __attribute__((ext_vector_type(4)))  int      i32x4;

#define NODES 10000
#define EDGES 160000
#define ETOT  (EDGES + NODES)
#define FIN   256
#define HEADS 8
#define CH    64
#define FHID  512
#define NCLS  40
#define NCLSP 64
#define MP    10112          // padded rows: multiple of 128 and 64
#define NEG_SLOPE 0.2f

#define CDIV(a, b) (((a) + (b) - 1) / (b))

// ---------------------------------------------------------------------------
// f32 -> f16 staging with zero padding (row-major copy)
// ---------------------------------------------------------------------------
__global__ void cvt_pad_f16(const float* __restrict__ s, _Float16* __restrict__ d,
                            int rows, int cols, int prows, int pcols) {
  int t = blockIdx.x * blockDim.x + threadIdx.x;
  if (t >= prows * pcols) return;
  int r = t / pcols, c = t % pcols;
  d[t] = (r < rows && c < cols) ? (_Float16)s[(size_t)r * cols + c] : (_Float16)0.f;
}

// f32 [rows x cols] -> f16 transposed [prows x pcols] = [Np x K], zero padded
__global__ void cvt_pad_f16_T(const float* __restrict__ s, _Float16* __restrict__ d,
                              int rows, int cols, int prows, int pcols) {
  int t = blockIdx.x * blockDim.x + threadIdx.x;
  if (t >= prows * pcols) return;
  int n = t / pcols, k = t % pcols;  // output (n, k) = input (k, n)
  d[t] = (n < cols && k < rows) ? (_Float16)s[(size_t)k * cols + n] : (_Float16)0.f;
}

__global__ void zero_f32(float* __restrict__ p, int n) {
  int t = blockIdx.x * blockDim.x + threadIdx.x;
  if (t < n) p[t] = 0.f;
}

// ---------------------------------------------------------------------------
// TDM: async 2D tile load global->LDS.
//   tile: tileRows rows x 32 f16 (64B) per row, row stride ldK elements.
//   LDS padding: +8B after every 64B row  ->  36-half (72B) LDS row stride.
// D# layout per CDNA5 ISA 8.3/8.4 (group0 128b, group1 256b; groups 2/3 zero).
// ---------------------------------------------------------------------------
__device__ __forceinline__ unsigned lds_off(const void* p) {
  // flat LDS address: [63:32] aperture, [31:0] byte offset within LDS
  return (unsigned)(unsigned long long)p;
}

__device__ __forceinline__ void tdm_load_tile(unsigned lds_byte, const _Float16* g,
                                              int tileRows, int ldK, int tensRows) {
  unsigned long long va = (unsigned long long)g;
  u32x4 g0;
  g0.x = 1u;                                             // count=1, user mode
  g0.y = lds_byte;                                       // lds_addr
  g0.z = (unsigned)va;                                   // global_addr[31:0]
  g0.w = (unsigned)((va >> 32) & 0x01FFFFFFull) | (2u << 30);  // addr[56:32], type=2
  i32x8 g1;
  // data_size=1 (2B), pad_enable, pad_interval=3 (16 dwords = 64B), pad_amount=1 (2 dwords = 8B)
  g1[0] = (1 << 16) | (1 << 20) | (3 << 22) | (1 << 25);
  g1[1] = (ldK & 0xFFFF) << 16;                          // tensor_dim0[15:0] @ bits 63:48
  g1[2] = ((ldK >> 16) & 0xFFFF) | ((tensRows & 0xFFFF) << 16);   // dim0 hi | dim1 lo
  g1[3] = ((tensRows >> 16) & 0xFFFF) | (32 << 16);      // dim1 hi | tile_dim0 = 32
  g1[4] = tileRows & 0xFFFF;                             // tile_dim1 | tile_dim2=0
  g1[5] = ldK;                                           // tensor_dim0_stride[31:0]
  g1[6] = 0;                                             // stride0 hi | stride1 lo
  g1[7] = 0;
  i32x4 z4 = {0, 0, 0, 0};
  i32x8 z8 = {0, 0, 0, 0, 0, 0, 0, 0};
  // clang-23 / therock-10.0 form: 6 args (g0, g1, g2, g3, extra, cpol)
  __builtin_amdgcn_tensor_load_to_lds(g0, g1, z4, z4, z8, 0);
}

// ---------------------------------------------------------------------------
// WMMA GEMM: C[M x NN] (f32) = A[M x K] (f16, row major) * BT[NN x K]^T (f16)
// 8 waves/block; wave covers 16(M) x 64(N) via 4 WMMA tiles; TK=32.
// TDM double-buffered LDS staging, fragments loaded as packed b32 per ISA 7.12.2.
// ---------------------------------------------------------------------------
template <int TM, int TN, int WM, int WN>
__global__ __launch_bounds__(256) void gemm_f16_wmma(
    const _Float16* __restrict__ A, const _Float16* __restrict__ BT,
    float* __restrict__ C, int M, int K, int NN) {
  constexpr int TK = 32;
  constexpr int LS = 36;  // halfs per LDS row (64B data + 8B TDM pad)
  __shared__ _Float16 lA[2][TM * LS];
  __shared__ _Float16 lB[2][TN * LS];

  const int tid  = threadIdx.x;
  const int lane = tid & 31;
  const int wave = tid >> 5;
  const int wm = wave % WM, wn = wave / WM;
  const int m0 = blockIdx.y * TM, n0 = blockIdx.x * TN;

  v8f acc[4] = {};

  if (wave == 0) {  // prologue: buffer 0
    tdm_load_tile(lds_off(&lA[0][0]), A  + (size_t)m0 * K, TM, K, M - m0);
    tdm_load_tile(lds_off(&lB[0][0]), BT + (size_t)n0 * K, TN, K, NN - n0);
  }

  for (int k0 = 0; k0 < K; k0 += TK) {
    const int cur = (k0 / TK) & 1;
    if (wave == 0) {
      if (k0 + TK < K) {
        tdm_load_tile(lds_off(&lA[cur ^ 1][0]), A  + (size_t)m0 * K + (k0 + TK), TM, K, M - m0);
        tdm_load_tile(lds_off(&lB[cur ^ 1][0]), BT + (size_t)n0 * K + (k0 + TK), TN, K, NN - n0);
        __builtin_amdgcn_s_wait_tensorcnt(2);  // current pair done, next pair in flight
      } else {
        __builtin_amdgcn_s_wait_tensorcnt(0);
      }
    }
    __syncthreads();

    // A fragment (16x32 f16 layout, ISA 7.12.2)
    union { v16h v; unsigned u[8]; } af;
    {
      int row = wm * 16 + (lane & 15);
      int hi8 = (lane >> 4) * 8;
#pragma unroll
      for (int j = 0; j < 8; ++j) {
        int kk = ((j < 4) ? 2 * j : 16 + 2 * (j - 4)) + hi8;
        af.u[j] = *(const unsigned*)&lA[cur][row * LS + kk];
      }
    }
#pragma unroll
    for (int t = 0; t < 4; ++t) {
      union { v16h v; unsigned u[8]; } bf;
      int col = wn * 64 + t * 16 + (lane & 15);
      int kb  = (lane >> 4) * 8;
#pragma unroll
      for (int j = 0; j < 8; ++j)
        bf.u[j] = *(const unsigned*)&lB[cur][col * LS + kb + ((j < 4) ? 2 * j : 16 + 2 * (j - 4))];
      acc[t] = __builtin_amdgcn_wmma_f32_16x16x32_f16(
          false, af.v, false, bf.v, (short)0, acc[t], false, false);
    }
    __syncthreads();
  }

  // C/D layout: VGPR i -> row i + (lane>=16)*8, col = lane&15
  int rb = m0 + wm * 16 + (lane >> 4) * 8;
#pragma unroll
  for (int t = 0; t < 4; ++t) {
    int cb = n0 + wn * 64 + t * 16 + (lane & 15);
#pragma unroll
    for (int i = 0; i < 8; ++i)
      C[(size_t)(rb + i) * NN + cb] = acc[t][i];
  }
}

// ---------------------------------------------------------------------------
// Attention helpers
// ---------------------------------------------------------------------------
__device__ __forceinline__ float lrelu(float v) { return v > 0.f ? v : v * NEG_SLOPE; }

__device__ __forceinline__ void atomicMaxF(float* p, float v) {
  if (v >= 0.f) atomicMax((int*)p, __float_as_int(v));
  else          atomicMin((unsigned int*)p, __float_as_uint(v));
}

template <int H, int C>
__global__ void gat_prep(const float* __restrict__ hx, int ld,
                         const float* __restrict__ asrc, const float* __restrict__ adst,
                         float* __restrict__ als, float* __restrict__ ald,
                         float* __restrict__ mx, float* __restrict__ den, int n) {
  int t = blockIdx.x * blockDim.x + threadIdx.x;
  if (t >= n * H) return;
  int node = t / H, h = t % H;
  const float* p = hx + (size_t)node * ld + h * C;
  float s = 0.f, d = 0.f;
#pragma unroll 8
  for (int c = 0; c < C; ++c) {
    float v = p[c];
    s += v * asrc[h * C + c];
    d += v * adst[h * C + c];
  }
  als[t] = s;
  ald[t] = d;
  mx[t]  = -__builtin_inff();
  den[t] = 0.f;
}

template <int H>
__global__ void gat_edge_max(const int* __restrict__ src, const int* __restrict__ dst,
                             const float* __restrict__ als, const float* __restrict__ ald,
                             float* __restrict__ mx) {
  int e = blockIdx.x * blockDim.x + threadIdx.x;
  if (e >= ETOT) return;
  int s = e < EDGES ? src[e] : e - EDGES;
  int d = e < EDGES ? dst[e] : e - EDGES;
#pragma unroll
  for (int h = 0; h < H; ++h)
    atomicMaxF(&mx[(size_t)d * H + h], lrelu(als[(size_t)s * H + h] + ald[(size_t)d * H + h]));
}

template <int H>
__global__ void gat_edge_expsum(const int* __restrict__ src, const int* __restrict__ dst,
                                const float* __restrict__ als, const float* __restrict__ ald,
                                const float* __restrict__ mx,
                                float* __restrict__ ex, float* __restrict__ den) {
  int e = blockIdx.x * blockDim.x + threadIdx.x;
  if (e >= ETOT) return;
  int s = e < EDGES ? src[e] : e - EDGES;
  int d = e < EDGES ? dst[e] : e - EDGES;
#pragma unroll
  for (int h = 0; h < H; ++h) {
    float v = lrelu(als[(size_t)s * H + h] + ald[(size_t)d * H + h]);
    float t = __expf(v - mx[(size_t)d * H + h]);
    ex[(size_t)e * H + h] = t;
    atomicAdd(&den[(size_t)d * H + h], t);
  }
}

// wave-per-edge weighted scatter: out[dst] += alpha_h * hx[src], alpha hoisted per head
template <int H, int C>
__global__ void gat_edge_scatter(const int* __restrict__ src, const int* __restrict__ dst,
                                 const float* __restrict__ hx, int ldh,
                                 const float* __restrict__ ex, const float* __restrict__ den,
                                 float* __restrict__ out, int ldo) {
  int gt = blockIdx.x * blockDim.x + threadIdx.x;
  int e = gt >> 5, lane = gt & 31;
  if (e >= ETOT) return;
  int s = e < EDGES ? src[e] : e - EDGES;
  int d = e < EDGES ? dst[e] : e - EDGES;
  const float* hs = hx + (size_t)s * ldh;
  float* od = out + (size_t)d * ldo;
#pragma unroll
  for (int h = 0; h < H; ++h) {
    float alpha = ex[(size_t)e * H + h] / (den[(size_t)d * H + h] + 1e-16f);
    for (int c = lane; c < C; c += 32)
      atomicAdd(&od[h * C + c], alpha * hs[h * C + c]);
  }
}

// bias + ELU, emit f16 staging for the next GEMM (zeros in padded rows)
__global__ void gat_finalize(const float* __restrict__ outv, const float* __restrict__ bias,
                             _Float16* __restrict__ a16, int n, int cols, int prows) {
  int t = blockIdx.x * blockDim.x + threadIdx.x;
  if (t >= prows * cols) return;
  int r = t / cols, c = t % cols;
  float v = 0.f;
  if (r < n) {
    v = outv[(size_t)r * cols + c] + bias[c];
    v = v > 0.f ? v : expm1f(v);
  }
  a16[t] = (_Float16)v;
}

__global__ void bias_add(float* __restrict__ o, const float* __restrict__ b, int n, int cols) {
  int t = blockIdx.x * blockDim.x + threadIdx.x;
  if (t >= n * cols) return;
  o[t] += b[t % cols];
}

// ---------------------------------------------------------------------------
extern "C" void kernel_launch(void* const* d_in, const int* in_sizes, int n_in,
                              void* d_out, int out_size, void* d_ws, size_t ws_size,
                              hipStream_t stream) {
  const float* x   = (const float*)d_in[0];
  const int*   ei  = (const int*)  d_in[1];
  const float* W0  = (const float*)d_in[2];
  const float* as0 = (const float*)d_in[3];
  const float* ad0 = (const float*)d_in[4];
  const float* b0  = (const float*)d_in[5];
  const float* W1  = (const float*)d_in[6];
  const float* as1 = (const float*)d_in[7];
  const float* ad1 = (const float*)d_in[8];
  const float* b1  = (const float*)d_in[9];
  const float* W2  = (const float*)d_in[10];
  const float* as2 = (const float*)d_in[11];
  const float* ad2 = (const float*)d_in[12];
  const float* b2  = (const float*)d_in[13];
  float* out = (float*)d_out;

  char* w = (char*)d_ws;
  auto carve = [&](size_t bytes) {
    char* p = w;
    w += (bytes + 255) & ~(size_t)255;
    return p;
  };
  _Float16* A16 = (_Float16*)carve((size_t)MP * FHID * 2);
  _Float16* B16 = (_Float16*)carve((size_t)FHID * FHID * 2);  // holds W^T (Np x K)
  float* HX  = (float*)carve((size_t)MP * FHID * 4);
  float* OUT = (float*)carve((size_t)NODES * FHID * 4);
  float* ALS = (float*)carve((size_t)NODES * HEADS * 4);
  float* ALD = (float*)carve((size_t)NODES * HEADS * 4);
  float* MX  = (float*)carve((size_t)NODES * HEADS * 4);
  float* DEN = (float*)carve((size_t)NODES * HEADS * 4);
  float* EX  = (float*)carve((size_t)ETOT * HEADS * 4);

  const int* srcI = ei;
  const int* dstI = ei + EDGES;
  const int B = 256;

  // ------------------ layer 0 ------------------
  cvt_pad_f16<<<CDIV(MP * FIN, B), B, 0, stream>>>(x, A16, NODES, FIN, MP, FIN);
  cvt_pad_f16_T<<<CDIV(FHID * FIN, B), B, 0, stream>>>(W0, B16, FIN, FHID, FHID, FIN);
  gemm_f16_wmma<64, 128, 4, 2><<<dim3(FHID / 128, MP / 64), B, 0, stream>>>(
      A16, B16, HX, MP, FIN, FHID);
  zero_f32<<<CDIV(NODES * FHID, B), B, 0, stream>>>(OUT, NODES * FHID);
  gat_prep<HEADS, CH><<<CDIV(NODES * HEADS, B), B, 0, stream>>>(
      HX, FHID, as0, ad0, ALS, ALD, MX, DEN, NODES);
  gat_edge_max<HEADS><<<CDIV(ETOT, B), B, 0, stream>>>(srcI, dstI, ALS, ALD, MX);
  gat_edge_expsum<HEADS><<<CDIV(ETOT, B), B, 0, stream>>>(srcI, dstI, ALS, ALD, MX, EX, DEN);
  gat_edge_scatter<HEADS, CH><<<CDIV(ETOT * 32, B), B, 0, stream>>>(
      srcI, dstI, HX, FHID, EX, DEN, OUT, FHID);
  gat_finalize<<<CDIV(MP * FHID, B), B, 0, stream>>>(OUT, b0, A16, NODES, FHID, MP);

  // ------------------ layer 1 ------------------
  cvt_pad_f16_T<<<CDIV(FHID * FHID, B), B, 0, stream>>>(W1, B16, FHID, FHID, FHID, FHID);
  gemm_f16_wmma<64, 128, 4, 2><<<dim3(FHID / 128, MP / 64), B, 0, stream>>>(
      A16, B16, HX, MP, FHID, FHID);
  zero_f32<<<CDIV(NODES * FHID, B), B, 0, stream>>>(OUT, NODES * FHID);
  gat_prep<HEADS, CH><<<CDIV(NODES * HEADS, B), B, 0, stream>>>(
      HX, FHID, as1, ad1, ALS, ALD, MX, DEN, NODES);
  gat_edge_max<HEADS><<<CDIV(ETOT, B), B, 0, stream>>>(srcI, dstI, ALS, ALD, MX);
  gat_edge_expsum<HEADS><<<CDIV(ETOT, B), B, 0, stream>>>(srcI, dstI, ALS, ALD, MX, EX, DEN);
  gat_edge_scatter<HEADS, CH><<<CDIV(ETOT * 32, B), B, 0, stream>>>(
      srcI, dstI, HX, FHID, EX, DEN, OUT, FHID);
  gat_finalize<<<CDIV(MP * FHID, B), B, 0, stream>>>(OUT, b1, A16, NODES, FHID, MP);

  // ------------------ layer 2 (H=1, C=40, cols padded to 64) ------------------
  cvt_pad_f16_T<<<CDIV(NCLSP * FHID, B), B, 0, stream>>>(W2, B16, FHID, NCLS, NCLSP, FHID);
  gemm_f16_wmma<128, 64, 8, 1><<<dim3(NCLSP / 64, MP / 128), B, 0, stream>>>(
      A16, B16, HX, MP, FHID, NCLSP);
  zero_f32<<<CDIV(NODES * NCLS, B), B, 0, stream>>>(out, NODES * NCLS);
  gat_prep<1, NCLS><<<CDIV(NODES, B), B, 0, stream>>>(
      HX, NCLSP, as2, ad2, ALS, ALD, MX, DEN, NODES);
  gat_edge_max<1><<<CDIV(ETOT, B), B, 0, stream>>>(srcI, dstI, ALS, ALD, MX);
  gat_edge_expsum<1><<<CDIV(ETOT, B), B, 0, stream>>>(srcI, dstI, ALS, ALD, MX, EX, DEN);
  gat_edge_scatter<1, NCLS><<<CDIV(ETOT * 32, B), B, 0, stream>>>(
      srcI, dstI, HX, NCLSP, EX, DEN, out, NCLS);
  bias_add<<<CDIV(NODES * NCLS, B), B, 0, stream>>>(out, b2, NODES, NCLS);
}
